// ReconBlock_63591285785186
// MI455X (gfx1250) — matrix operand order repaired and checked by reading, not compiled, not run on hardware
//
#include <hip/hip_runtime.h>
#include <hip/hip_bf16.h>

#define NVOX 262144
#define CCH  128
#define EPSB 1e-3f

typedef __attribute__((ext_vector_type(16))) _Float16 v16h;
typedef __attribute__((ext_vector_type(8)))  _Float16 v8h;
typedef __attribute__((ext_vector_type(8)))  float    v8f;

// ---------------- Prep 1: features fp32 -> fp16, plus zeroed sentinel row ----
__global__ void cvt_feats_kernel(const float* __restrict__ f, _Float16* __restrict__ o) {
  long idx = (long)blockIdx.x * 256 + threadIdx.x;
  const long total = (long)(NVOX + 1) * CCH;
  if (idx >= total) return;
  o[idx] = (idx < (long)NVOX * CCH) ? (_Float16)f[idx] : (_Float16)0.0f;
}

// ---------------- Prep 2: pack W[3][3][128][128] into WMMA-B fragments -------
// Fragment layout (per (br,tap,kb,nb)): 512 halfs =
//   chunk0: lane*8 halfs (B elements 0..7), chunk1 at +256: elements 8..15.
// Lane l holds column n = l%16, rows K = kb*32 + (l<16?0:8) + {0..7, 16..23}.
__global__ void pack_w_kernel(const float* __restrict__ W1, const float* __restrict__ W2,
                              const float* __restrict__ W3, _Float16* __restrict__ wp) {
  int idx = blockIdx.x * 256 + threadIdx.x;
  const int total = 3 * 3 * 4 * 8 * 512;   // 147456
  if (idx >= total) return;
  int j     = idx & 7;
  int lane  = (idx >> 3) & 31;
  int chunk = (idx >> 8) & 1;
  int nb    = (idx >> 9) & 7;
  int kb    = (idx >> 12) & 3;
  int tap   = (idx >> 14) % 3;
  int br    = idx / 49152;
  int jj = chunk * 8 + j;
  int k  = kb * 32 + ((lane < 16) ? 0 : 8) + chunk * 16 + j;   // input channel
  int c  = nb * 16 + (lane & 15);                   // output channel
  const float* W = (br == 0) ? W1 : ((br == 1) ? W2 : W3);
  wp[idx] = (_Float16)W[((size_t)tap * CCH + k) * CCH + c];
}

// ---------------- Main fused kernel -----------------------------------------
__global__ __launch_bounds__(256)
void subm_fused_kernel(const _Float16* __restrict__ feats16,
                       const _Float16* __restrict__ wpack,
                       const float* __restrict__ featf,
                       const int* __restrict__ nbrZ,
                       const int* __restrict__ nbrY,
                       const int* __restrict__ nbrX,
                       const float* __restrict__ g0, const float* __restrict__ b0,
                       const float* __restrict__ m0, const float* __restrict__ v0,
                       const float* __restrict__ g2, const float* __restrict__ b2,
                       const float* __restrict__ m2, const float* __restrict__ v2,
                       const float* __restrict__ g3, const float* __restrict__ b3,
                       const float* __restrict__ m3, const float* __restrict__ v3,
                       float* __restrict__ out)
{
  const int lane  = threadIdx.x & 31;
  const int wave  = threadIdx.x >> 5;
  const int tile  = blockIdx.x * 8 + wave;   // 16-voxel tile per wave
  const int vbase = tile * 16;
  const int mrow  = lane & 15;               // A-matrix row (voxel within tile)
  const int h8    = (lane >= 16) ? 8 : 0;    // A layout half-select
  const v8f vzero = {};

  // One branch's packed weight fragments: 3 taps x 4 kblocks x 8 ntiles x 512 halfs = 96 KB
  __shared__ alignas(16) _Float16 ldsW[3 * 4 * 8 * 512];

  // Cooperative stage of branch br's weights into LDS (96 KB / block, 16B units)
  auto stage = [&](int br) {
    const uint4* __restrict__ src = (const uint4*)(wpack + (size_t)br * 49152);
    uint4* dst = (uint4*)ldsW;
#pragma unroll 4
    for (int i = threadIdx.x; i < 6144; i += 256) dst[i] = src[i];
  };

  v8f sum[8];
  v8f acc[8];
#pragma unroll
  for (int i = 0; i < 8; ++i) sum[i] = vzero;

  // K-loop for one branch: acc[0..7] += sum_{tap,k} A(gathered feats) * B(ldsW)
  auto mm = [&](const int* __restrict__ nbt) {
#pragma unroll
    for (int i = 0; i < 8; ++i) acc[i] = vzero;
    for (int tap = 0; tap < 3; ++tap) {
      int row = nbt[(vbase + mrow) * 3 + tap];          // sentinel NVOX -> zero row
      const _Float16* rp = feats16 + (size_t)row * CCH;
#pragma unroll
      for (int kb = 0; kb < 4; ++kb) {
        // 16-bit A 16x32 layout: lane<16 holds K {kb*32+0..7, +16..23}, lane>=16 the +8 halves
        v8h alo = *(const v8h*)(rp + kb * 32 + h8);
        v8h ahi = *(const v8h*)(rp + kb * 32 + 16 + h8);
        v16h a;
#pragma unroll
        for (int i = 0; i < 8; ++i) { a[i] = alo[i]; a[8 + i] = ahi[i]; }
        const _Float16* wf = ldsW + ((tap * 4 + kb) * 8) * 512 + lane * 8;
#pragma unroll
        for (int nb = 0; nb < 8; ++nb) {
          v8h blo = *(const v8h*)(wf + nb * 512);
          v8h bhi = *(const v8h*)(wf + nb * 512 + 256);
          v16h b;
#pragma unroll
          for (int i = 0; i < 8; ++i) { b[i] = blo[i]; b[8 + i] = bhi[i]; }
          acc[nb] = __builtin_amdgcn_wmma_f32_16x16x32_f16(
              false, a, false, b, (short)0, acc[nb], false, false);
        }
      }
    }
  };

  // ---- Branch 1: conv3x1x1 -> bn0 -> sigmoid -> bn2 -> sigmoid ----
  stage(0);
  __syncthreads();
  mm(nbrZ);
#pragma unroll
  for (int nb = 0; nb < 8; ++nb) {
    int c = nb * 16 + mrow;                  // output channel of this lane/tile
    float s0 = g0[c] * rsqrtf(v0[c] + EPSB);
    float h0 = b0[c] - m0[c] * s0;
    float s2 = g2[c] * rsqrtf(v2[c] + EPSB);
    float h2 = b2[c] - m2[c] * s2;
#pragma unroll
    for (int i = 0; i < 8; ++i) {
      float x = acc[nb][i] * s0 + h0;
      x = 1.0f / (1.0f + __expf(-x));
      x = x * s2 + h2;
      x = 1.0f / (1.0f + __expf(-x));
      sum[nb][i] += x;
    }
  }

  // ---- Branch 2: conv1x3x1, added raw ----
  __syncthreads();
  stage(1);
  __syncthreads();
  mm(nbrY);
#pragma unroll
  for (int nb = 0; nb < 8; ++nb)
#pragma unroll
    for (int i = 0; i < 8; ++i) sum[nb][i] += acc[nb][i];

  // ---- Branch 3: conv1x1x3 -> bn3 -> sigmoid ----
  __syncthreads();
  stage(2);
  __syncthreads();
  mm(nbrX);
#pragma unroll
  for (int nb = 0; nb < 8; ++nb) {
    int c = nb * 16 + mrow;
    float s3 = g3[c] * rsqrtf(v3[c] + EPSB);
    float h3 = b3[c] - m3[c] * s3;
#pragma unroll
    for (int i = 0; i < 8; ++i) {
      float x = acc[nb][i] * s3 + h3;
      x = 1.0f / (1.0f + __expf(-x));
      sum[nb][i] += x;
    }
  }

  // ---- (s + s2 + s3) * features, store fp32 ----
  // C/D layout: lane<16 -> row = vgpr i, col = lane; lane>=16 -> row = i+8, col = lane-16
  const int rofs = (lane >= 16) ? 8 : 0;
#pragma unroll
  for (int nb = 0; nb < 8; ++nb) {
    int c = nb * 16 + mrow;
#pragma unroll
    for (int i = 0; i < 8; ++i) {
      size_t off = (size_t)(vbase + i + rofs) * CCH + c;
      out[off] = sum[nb][i] * featf[off];
    }
  }
}

extern "C" void kernel_launch(void* const* d_in, const int* in_sizes, int n_in,
                              void* d_out, int out_size, void* d_ws, size_t ws_size,
                              hipStream_t stream) {
  const float* features = (const float*)d_in[0];
  const float* W1 = (const float*)d_in[1];
  const float* W2 = (const float*)d_in[2];
  const float* W3 = (const float*)d_in[3];
  const float* g0 = (const float*)d_in[4];
  const float* b0 = (const float*)d_in[5];
  const float* m0 = (const float*)d_in[6];
  const float* v0 = (const float*)d_in[7];
  const float* g2 = (const float*)d_in[8];
  const float* b2 = (const float*)d_in[9];
  const float* m2 = (const float*)d_in[10];
  const float* v2 = (const float*)d_in[11];
  const float* g3 = (const float*)d_in[12];
  const float* b3 = (const float*)d_in[13];
  const float* m3 = (const float*)d_in[14];
  const float* v3 = (const float*)d_in[15];
  const int* nbr_z = (const int*)d_in[16];
  const int* nbr_y = (const int*)d_in[17];
  const int* nbr_x = (const int*)d_in[18];

  _Float16* feats16 = (_Float16*)d_ws;                       // (N+1) x 128 fp16
  _Float16* wpack   = feats16 + (size_t)(NVOX + 1) * CCH;    // 147456 fp16

  {
    long total = (long)(NVOX + 1) * CCH;
    int blocks = (int)((total + 255) / 256);
    cvt_feats_kernel<<<blocks, 256, 0, stream>>>(features, feats16);
  }
  {
    int total = 3 * 3 * 4 * 8 * 512;
    pack_w_kernel<<<(total + 255) / 256, 256, 0, stream>>>(W1, W2, W3, wpack);
  }
  {
    int blocks = NVOX / 16 / 8;   // one 16-voxel tile per wave, 8 waves/block
    subm_fused_kernel<<<blocks, 256, 0, stream>>>(
        feats16, wpack, features, nbr_z, nbr_y, nbr_x,
        g0, b0, m0, v0, g2, b2, m2, v2, g3, b3, m3, v3,
        (float*)d_out);
  }
}
